// Block_47493748359290
// MI455X (gfx1250) — matrix operand order repaired and checked by reading, not compiled, not run on hardware
//
#include <hip/hip_runtime.h>

// ---------------------------------------------------------------------------
// Types / helpers
// ---------------------------------------------------------------------------
typedef __attribute__((ext_vector_type(16))) __bf16 v16bf;
typedef __attribute__((ext_vector_type(8)))  float  v8f;
typedef unsigned short u16;
typedef __attribute__((ext_vector_type(8))) u16 u16x8;
typedef __attribute__((ext_vector_type(4))) unsigned int u32x4;
typedef __attribute__((ext_vector_type(8))) int i32x8;
typedef __attribute__((ext_vector_type(4))) int i32x4;

union FragU { v16bf bf; struct { u16x8 lo, hi; } h; };

__device__ __forceinline__ u16 f2bf(float f) {
  union { float f; unsigned u; } c; c.f = f;
  unsigned r = (c.u + 0x7FFFu + ((c.u >> 16) & 1u)) >> 16;
  return (u16)r;
}
__device__ __forceinline__ v8f vzero() {
  v8f v = {0.f,0.f,0.f,0.f,0.f,0.f,0.f,0.f}; return v;
}
__device__ __forceinline__ float halfmax(float v) {
  v = fmaxf(v, __shfl_xor(v, 1, 32));
  v = fmaxf(v, __shfl_xor(v, 2, 32));
  v = fmaxf(v, __shfl_xor(v, 4, 32));
  v = fmaxf(v, __shfl_xor(v, 8, 32));
  return v;
}
__device__ __forceinline__ float halfsum(float v) {
  v += __shfl_xor(v, 1, 32);
  v += __shfl_xor(v, 2, 32);
  v += __shfl_xor(v, 4, 32);
  v += __shfl_xor(v, 8, 32);
  return v;
}

// ---------------------------------------------------------------------------
// Tensor Data Mover: 2D tile (2-byte elements) global -> LDS, with LDS row
// padding.  D# packing per CDNA5 ISA.
//   pad_interval code: pad after 2<<code DWORDs; pad_amount code: (code+1) DW.
// ---------------------------------------------------------------------------
#if __has_builtin(__builtin_amdgcn_tensor_load_to_lds)
#define HAVE_TDM 1
#else
#define HAVE_TDM 0
#endif

#if HAVE_TDM
__device__ __forceinline__ void tdm_load_2d(unsigned lds_off, const void* gptr,
                                            unsigned tensor_d0,
                                            unsigned tensor_d1,
                                            unsigned tile_d0, unsigned tile_d1,
                                            unsigned long long stride0_elems,
                                            unsigned pad_interval_code,
                                            unsigned pad_amount_code) {
  unsigned long long ga = (unsigned long long)gptr;
  u32x4 g0;
  g0[0] = 1u;                                   // count=1, user-mode D#
  g0[1] = lds_off;                              // lds_addr (bytes)
  g0[2] = (unsigned)(ga & 0xffffffffu);         // global_addr[31:0]
  g0[3] = (unsigned)((ga >> 32) & 0x01ffffffu)  // global_addr[56:32]
        | (2u << 30);                           // type = 2 ("image")
  i32x8 g1;
  unsigned d0 = (1u << 16)                      // data_size = 2 bytes
              | (1u << 20)                      // pad_enable
              | (pad_interval_code << 22) | (pad_amount_code << 25);
  g1[0] = (int)d0;
  g1[1] = (int)((tensor_d0 & 0xffffu) << 16);                      // dim0 lo
  g1[2] = (int)(((tensor_d0 >> 16) & 0xffffu) |
                ((tensor_d1 & 0xffffu) << 16));                    // dim0 hi|dim1 lo
  g1[3] = (int)(((tensor_d1 >> 16) & 0xffffu) |
                ((tile_d0 & 0xffffu) << 16));                      // dim1 hi|tile0
  g1[4] = (int)(tile_d1 & 0xffffu);                                // tile1, tile2=0
  g1[5] = (int)(unsigned)(stride0_elems & 0xffffffffu);            // stride0 lo
  g1[6] = (int)(unsigned)((stride0_elems >> 32) & 0xffffu);        // stride0 hi
  g1[7] = 0;
  i32x4 gz = {0, 0, 0, 0};
#if defined(__clang_major__) && __clang_major__ >= 23
  i32x8 gz8 = {0, 0, 0, 0, 0, 0, 0, 0};
  __builtin_amdgcn_tensor_load_to_lds(g0, g1, gz, gz, gz8, 0);
#else
  __builtin_amdgcn_tensor_load_to_lds(g0, g1, gz, gz, 0);
#endif
}
#endif

// ---------------------------------------------------------------------------
// fp32 -> bf16 conversion (weights, done once)
// ---------------------------------------------------------------------------
__global__ __launch_bounds__(256) void cvt_kernel(const float* __restrict__ src,
                                                  u16* __restrict__ dst, int n) {
  int i = blockIdx.x * 256 + threadIdx.x;
  if (i < n) dst[i] = f2bf(src[i]);
}

// ---------------------------------------------------------------------------
// LayerNorm: one wave per token (E = 384 = 32 lanes * 12), bf16 out
// ---------------------------------------------------------------------------
__global__ __launch_bounds__(256) void ln_kernel(const float* __restrict__ x,
                                                 const float* __restrict__ g,
                                                 const float* __restrict__ be,
                                                 u16* __restrict__ out) {
  const int lane = threadIdx.x & 31;
  const int wid  = threadIdx.x >> 5;
  const size_t token = (size_t)blockIdx.x * 8 + wid;
  const float* xr = x + token * 384;
  float vals[12];
  float s = 0.f, sq = 0.f;
#pragma unroll
  for (int i = 0; i < 12; ++i) {
    float v = xr[lane + i * 32];
    vals[i] = v; s += v; sq += v * v;
  }
#pragma unroll
  for (int m = 16; m >= 1; m >>= 1) {
    s  += __shfl_xor(s,  m, 32);
    sq += __shfl_xor(sq, m, 32);
  }
  const float mean = s * (1.f / 384.f);
  const float var  = sq * (1.f / 384.f) - mean * mean;
  const float rstd = rsqrtf(var + 1e-5f);
  u16* orow = out + token * 384;
#pragma unroll
  for (int i = 0; i < 12; ++i) {
    int col = lane + i * 32;
    orow[col] = f2bf((vals[i] - mean) * rstd * g[col] + be[col]);
  }
}

// ---------------------------------------------------------------------------
// bf16 WMMA GEMM:  C[M,N] = A[M,K](bf16, row-major) * W[K,N](bf16, row-major)
// Block tile 128x64, K-chunk 32, 8 waves -> 4x2 grid of 32x32 wave tiles.
// Double-buffered LDS; A tile via TDM (wave 0), B tile manual (transposed).
// ---------------------------------------------------------------------------
enum { MODE_QK = 0, MODE_V = 1, MODE_FRES = 2, MODE_RELU = 3 };

#define BM 128
#define BN 64
#define BK 32
#define ASTR 40   // ushort stride, 80B (mult of 16 -> b128-aligned frags)
#define BSTR 40

__global__ __launch_bounds__(256) void gemm_bf16_kernel(
    const u16* __restrict__ A, const u16* __restrict__ W,
    int M, int N, int K, int mode,
    const float* __restrict__ bias, const float* __restrict__ resid,
    float* __restrict__ outf, u16* __restrict__ outb) {
  __shared__ __align__(16) u16 sA[2][BM * ASTR];
  __shared__ __align__(16) u16 sB[2][BN * BSTR];   // transposed: [n][k]

  const int tid  = threadIdx.x;
  const int lane = tid & 31, wid = tid >> 5;
  const int half = lane >> 4, l16 = lane & 15;
  const int m0 = blockIdx.x * BM;
  const int n0 = blockIdx.y * BN;
  const int wm = (wid & 3) * 32;
  const int wn = (wid >> 2) * 32;
  const int kchunks = K / BK;

  v8f acc[2][2];
#pragma unroll
  for (int i = 0; i < 2; ++i)
#pragma unroll
    for (int j = 0; j < 2; ++j) acc[i][j] = vzero();

  auto stageA = [&](int kc, int buf) {
#if HAVE_TDM
    if (wid == 0)   // one wave drives the DMA; tile 32x128, LDS stride 40
      tdm_load_2d((unsigned)(size_t)&sA[buf][0],
                  A + (size_t)m0 * K + (size_t)kc * BK,
                  (unsigned)K, (unsigned)M, BK, BM, (unsigned long long)K,
                  /*pad: 16 DW rows*/ 3u, /*pad 4 DW*/ 3u);
#else
#pragma unroll
    for (int i = 0; i < 4; ++i) {
      int gidx = tid + i * 256;
      int row = gidx >> 3, cg = gidx & 7;
      *(uint2*)&sA[buf][row * ASTR + cg * 4] =
          *(const uint2*)&A[(size_t)(m0 + row) * K + kc * BK + cg * 4];
    }
#endif
  };
  auto stageB = [&](int kc, int buf) {
#pragma unroll
    for (int i = 0; i < 8; ++i) {
      int e = tid + i * 256;
      int nn = e & 63, kk = e >> 6;
      sB[buf][nn * BSTR + kk] = W[(size_t)(kc * BK + kk) * N + n0 + nn];
    }
    if (kc + 1 < kchunks)   // prefetch next B strip into caches
      __builtin_prefetch(&W[(size_t)((kc + 1) * BK + (tid >> 6)) * N + n0 +
                            (tid & 63)], 0, 1);
  };
  auto waitStage = [&]() {
#if HAVE_TDM
    if (wid == 0) __builtin_amdgcn_s_wait_tensorcnt((short)0);
#endif
    __syncthreads();
  };

  stageA(0, 0);
  stageB(0, 0);
  waitStage();

  for (int kc = 0; kc < kchunks; ++kc) {
    const int cur = kc & 1, nxt = cur ^ 1;
    if (kc + 1 < kchunks) {      // overlap next tile DMA with this chunk's MMA
      stageA(kc + 1, nxt);
      stageB(kc + 1, nxt);
    }
    FragU af[2], bfm[2];
#pragma unroll
    for (int rt = 0; rt < 2; ++rt) {
      int row = wm + rt * 16 + l16;
      int kb = half * 8;               // A-frag: lanes 16-31 start at K=8
      af[rt].h.lo = *(const u16x8*)&sA[cur][row * ASTR + kb];
      af[rt].h.hi = *(const u16x8*)&sA[cur][row * ASTR + kb + 16];
    }
#pragma unroll
    for (int ct = 0; ct < 2; ++ct) {
      int col = wn + ct * 16 + l16;
      int kb = half * 8;               // B-frag: lanes 16-31 start at K=8
      bfm[ct].h.lo = *(const u16x8*)&sB[cur][col * BSTR + kb];
      bfm[ct].h.hi = *(const u16x8*)&sB[cur][col * BSTR + kb + 16];
    }
#pragma unroll
    for (int rt = 0; rt < 2; ++rt)
#pragma unroll
      for (int ct = 0; ct < 2; ++ct)
        acc[rt][ct] = __builtin_amdgcn_wmma_f32_16x16x32_bf16(
            false, af[rt].bf, false, bfm[ct].bf, (short)0, acc[rt][ct],
            false, false);
    if (kc + 1 < kchunks) waitStage();
  }

  // epilogue: C-frag layout -> row = r + 8*half, col = l16
#pragma unroll
  for (int rt = 0; rt < 2; ++rt)
#pragma unroll
    for (int ct = 0; ct < 2; ++ct)
#pragma unroll
      for (int r = 0; r < 8; ++r) {
        const int row = m0 + wm + rt * 16 + r + half * 8;
        const int col = n0 + wn + ct * 16 + l16;
        float v = acc[rt][ct][r];
        if (mode == MODE_QK) {
          int bb = row >> 8, t = row & 255;
          int hh = col >> 6, hs = col & 63;
          outb[((size_t)(bb * 6 + hh) * 256 + t) * 64 + hs] = f2bf(v);
        } else if (mode == MODE_V) {     // store V transposed: [B,H,HS,T]
          int bb = row >> 8, t = row & 255;
          int hh = col >> 6, hs = col & 63;
          outb[((size_t)(bb * 6 + hh) * 64 + hs) * 256 + t] = f2bf(v);
        } else if (mode == MODE_FRES) {
          v += bias[col] + resid[(size_t)row * N + col];
          outf[(size_t)row * N + col] = v;
        } else {                         // MODE_RELU
          v += bias[col];
          v = v > 0.f ? v : 0.f;
          outb[(size_t)row * N + col] = f2bf(v);
        }
      }
}

// ---------------------------------------------------------------------------
// Causal flash attention, one workgroup per (b, head).
// q,k in LDS as [T][HS] (stride 72); v in LDS transposed [HS][T] (stride 264).
// Tiles staged by TDM (waves 0..2).  Wave w owns query rows [32w, 32w+32).
// ---------------------------------------------------------------------------
__global__ __launch_bounds__(256) void attn_kernel(
    const u16* __restrict__ qg, const u16* __restrict__ kg,
    const u16* __restrict__ vg, u16* __restrict__ outb) {
  __shared__ __align__(16) u16 qs[256 * 72];
  __shared__ __align__(16) u16 ks[256 * 72];
  __shared__ __align__(16) u16 vt[64 * 264];
  __shared__ __align__(16) u16 pb[8][32 * 40];   // per-wave P staging

  const int tid  = threadIdx.x;
  const int lane = tid & 31, w = tid >> 5;
  const int half = lane >> 4, l16 = lane & 15;
  const int bh = blockIdx.x;
  const int bb = bh / 6, hh = bh % 6;
  const u16* q = qg + (size_t)bh * 256 * 64;
  const u16* k = kg + (size_t)bh * 256 * 64;
  const u16* v = vg + (size_t)bh * 64 * 256;

#if HAVE_TDM
  // q/k rows: 64 elems (32 DW) + 4 DW pad -> stride 72; v rows: 256 elems
  // (128 DW) + 4 DW pad -> stride 264.
  if (w == 0)
    tdm_load_2d((unsigned)(size_t)&qs[0], q, 64u, 256u, 64u, 256u, 64ull,
                4u, 3u);
  else if (w == 1)
    tdm_load_2d((unsigned)(size_t)&ks[0], k, 64u, 256u, 64u, 256u, 64ull,
                4u, 3u);
  else if (w == 2)
    tdm_load_2d((unsigned)(size_t)&vt[0], v, 256u, 64u, 256u, 64u, 256ull,
                6u, 3u);
  if (w < 3) __builtin_amdgcn_s_wait_tensorcnt((short)0);
#else
#pragma unroll
  for (int i = 0; i < 16; ++i) {
    int gidx = tid + i * 256;
    int t = gidx >> 4, hsg = gidx & 15;
    *(uint2*)&qs[t * 72 + hsg * 4] = *(const uint2*)&q[t * 64 + hsg * 4];
    *(uint2*)&ks[t * 72 + hsg * 4] = *(const uint2*)&k[t * 64 + hsg * 4];
  }
#pragma unroll
  for (int i = 0; i < 16; ++i) {
    int gidx = tid + i * 256;
    int hs = gidx >> 6, tg = gidx & 63;
    *(uint2*)&vt[hs * 264 + tg * 4] = *(const uint2*)&v[hs * 256 + tg * 4];
  }
#endif
  __syncthreads();

  const int m0 = w * 32;
  v8f co[2][4];
  float mstat[2][8], lstat[2][8];
#pragma unroll
  for (int rt = 0; rt < 2; ++rt) {
#pragma unroll
    for (int c = 0; c < 4; ++c) co[rt][c] = vzero();
#pragma unroll
    for (int r = 0; r < 8; ++r) { mstat[rt][r] = -1e30f; lstat[rt][r] = 0.f; }
  }

  for (int jb = 0; jb <= w; ++jb) {           // causal: only blocks <= w
    v8f s[2][2];
    s[0][0] = vzero(); s[0][1] = vzero(); s[1][0] = vzero(); s[1][1] = vzero();
#pragma unroll
    for (int kc = 0; kc < 2; ++kc) {          // HS = 64 -> two K=32 chunks
      FragU aq[2], bk[2];
#pragma unroll
      for (int rt = 0; rt < 2; ++rt) {
        int row = m0 + rt * 16 + l16;
        int kb = kc * 32 + half * 8;
        aq[rt].h.lo = *(const u16x8*)&qs[row * 72 + kb];
        aq[rt].h.hi = *(const u16x8*)&qs[row * 72 + kb + 16];
      }
#pragma unroll
      for (int ct = 0; ct < 2; ++ct) {
        int col = jb * 32 + ct * 16 + l16;    // B = k^T: col = key idx
        int kb = kc * 32 + half * 8;
        bk[ct].h.lo = *(const u16x8*)&ks[col * 72 + kb];
        bk[ct].h.hi = *(const u16x8*)&ks[col * 72 + kb + 16];
      }
#pragma unroll
      for (int rt = 0; rt < 2; ++rt)
#pragma unroll
        for (int ct = 0; ct < 2; ++ct)
          s[rt][ct] = __builtin_amdgcn_wmma_f32_16x16x32_bf16(
              false, aq[rt].bf, false, bk[ct].bf, (short)0, s[rt][ct],
              false, false);
    }
    // scale + causal mask (mask needed only on the diagonal block jb == w)
#pragma unroll
    for (int rt = 0; rt < 2; ++rt)
#pragma unroll
      for (int ct = 0; ct < 2; ++ct)
#pragma unroll
        for (int r = 0; r < 8; ++r) {
          float val = s[rt][ct][r] * 0.125f;  // HS^-0.5 = 1/8
          if (jb == w && (ct * 16 + l16) > (rt * 16 + r + half * 8))
            val = -1e30f;
          s[rt][ct][r] = val;
        }
    // online softmax update + stage P (bf16) to per-wave LDS
#pragma unroll
    for (int rt = 0; rt < 2; ++rt) {
      float alpha[8];
#pragma unroll
      for (int r = 0; r < 8; ++r) {
        float rowmax = halfmax(fmaxf(s[rt][0][r], s[rt][1][r]));
        float mn = fmaxf(mstat[rt][r], rowmax);
        alpha[r] = __expf(mstat[rt][r] - mn);
        mstat[rt][r] = mn;
      }
#pragma unroll
      for (int ct = 0; ct < 2; ++ct)
#pragma unroll
        for (int r = 0; r < 8; ++r)
          s[rt][ct][r] = __expf(s[rt][ct][r] - mstat[rt][r]);
#pragma unroll
      for (int r = 0; r < 8; ++r) {
        float rowsum = halfsum(s[rt][0][r] + s[rt][1][r]);
        lstat[rt][r] = lstat[rt][r] * alpha[r] + rowsum;
      }
#pragma unroll
      for (int c = 0; c < 4; ++c)
#pragma unroll
        for (int r = 0; r < 8; ++r) co[rt][c][r] *= alpha[r];
#pragma unroll
      for (int ct = 0; ct < 2; ++ct)
#pragma unroll
        for (int r = 0; r < 8; ++r)
          pb[w][(rt * 16 + r + half * 8) * 40 + ct * 16 + l16] =
              f2bf(s[rt][ct][r]);
    }
    // wave-local LDS exchange: make stores visible before frag reload
    asm volatile("s_wait_dscnt 0" ::: "memory");
    FragU ap[2], bv[4];
#pragma unroll
    for (int rt = 0; rt < 2; ++rt) {
      int row = rt * 16 + l16;
      int kb = half * 8;
      ap[rt].h.lo = *(const u16x8*)&pb[w][row * 40 + kb];
      ap[rt].h.hi = *(const u16x8*)&pb[w][row * 40 + kb + 16];
    }
#pragma unroll
    for (int c = 0; c < 4; ++c) {
      int hs = c * 16 + l16;
      int kb = jb * 32 + half * 8;
      bv[c].h.lo = *(const u16x8*)&vt[hs * 264 + kb];
      bv[c].h.hi = *(const u16x8*)&vt[hs * 264 + kb + 16];
    }
#pragma unroll
    for (int rt = 0; rt < 2; ++rt)
#pragma unroll
      for (int c = 0; c < 4; ++c)
        co[rt][c] = __builtin_amdgcn_wmma_f32_16x16x32_bf16(
            false, ap[rt].bf, false, bv[c].bf, (short)0, co[rt][c],
            false, false);
    __builtin_amdgcn_wave_barrier();   // keep P reload ordered vs next store
  }

  // finalize: O /= l, write bf16 into attn [B,T,E] at column h*64+hs
#pragma unroll
  for (int rt = 0; rt < 2; ++rt)
#pragma unroll
    for (int r = 0; r < 8; ++r) {
      float inv = 1.f / lstat[rt][r];
      int t = m0 + rt * 16 + r + half * 8;
#pragma unroll
      for (int c = 0; c < 4; ++c) {
        int hs = c * 16 + l16;
        outb[((size_t)(bb * 256 + t)) * 384 + hh * 64 + hs] =
            f2bf(co[rt][c][r] * inv);
      }
    }
}

// ---------------------------------------------------------------------------
// Launcher
// ---------------------------------------------------------------------------
extern "C" void kernel_launch(void* const* d_in, const int* in_sizes, int n_in,
                              void* d_out, int out_size, void* d_ws,
                              size_t ws_size, hipStream_t stream) {
  (void)in_sizes; (void)n_in; (void)out_size; (void)ws_size;
  const float* x   = (const float*)d_in[0];
  const float* Wq  = (const float*)d_in[1];
  const float* Wk  = (const float*)d_in[2];
  const float* Wv  = (const float*)d_in[3];
  const float* Wo  = (const float*)d_in[4];
  const float* bo  = (const float*)d_in[5];
  const float* W1  = (const float*)d_in[6];
  const float* b1  = (const float*)d_in[7];
  const float* W2  = (const float*)d_in[8];
  const float* b2  = (const float*)d_in[9];
  const float* g1  = (const float*)d_in[10];
  const float* be1 = (const float*)d_in[11];
  const float* g2  = (const float*)d_in[12];
  const float* be2 = (const float*)d_in[13];

  char* ws = (char*)d_ws;
  const size_t SZ = 50331648;                 // 48 MB per bf16 [B,T,E] buffer
  u16*   qb    = (u16*)(ws + 0 * SZ);
  u16*   kb    = (u16*)(ws + 1 * SZ);
  u16*   vb    = (u16*)(ws + 2 * SZ);
  u16*   attnb = (u16*)(ws + 3 * SZ);
  float* y     = (float*)(ws + 0 * SZ);       // overlay: q,k dead post-attn
  u16*   ab    = (u16*)(ws + 0 * SZ);         // overlay: all dead post-LN2
  u16*   hb    = (u16*)(ws + 4 * SZ);
  u16*   Wqb   = (u16*)(ws + 5 * SZ);
  u16*   Wkb   = Wqb + 147456;
  u16*   Wvb   = Wkb + 147456;
  u16*   Wob   = Wvb + 147456;
  u16*   W1b   = Wob + 147456;
  u16*   W2b   = W1b + 589824;

  const int M = 65536;
  const int NEE = 147456, NE4 = 589824;
  cvt_kernel<<<(NEE + 255) / 256, 256, 0, stream>>>(Wq, Wqb, NEE);
  cvt_kernel<<<(NEE + 255) / 256, 256, 0, stream>>>(Wk, Wkb, NEE);
  cvt_kernel<<<(NEE + 255) / 256, 256, 0, stream>>>(Wv, Wvb, NEE);
  cvt_kernel<<<(NEE + 255) / 256, 256, 0, stream>>>(Wo, Wob, NEE);
  cvt_kernel<<<(NE4 + 255) / 256, 256, 0, stream>>>(W1, W1b, NE4);
  cvt_kernel<<<(NE4 + 255) / 256, 256, 0, stream>>>(W2, W2b, NE4);

  ln_kernel<<<8192, 256, 0, stream>>>(x, g1, be1, hb);

  dim3 g384(M / BM, 384 / BN), g1536(M / BM, 1536 / BN);
  gemm_bf16_kernel<<<g384, 256, 0, stream>>>(hb, Wqb, M, 384, 384, MODE_QK,
                                             nullptr, nullptr, nullptr, qb);
  gemm_bf16_kernel<<<g384, 256, 0, stream>>>(hb, Wkb, M, 384, 384, MODE_QK,
                                             nullptr, nullptr, nullptr, kb);
  gemm_bf16_kernel<<<g384, 256, 0, stream>>>(hb, Wvb, M, 384, 384, MODE_V,
                                             nullptr, nullptr, nullptr, vb);

  attn_kernel<<<1536, 256, 0, stream>>>(qb, kb, vb, attnb);

  gemm_bf16_kernel<<<g384, 256, 0, stream>>>(attnb, Wob, M, 384, 384,
                                             MODE_FRES, bo, x, y, nullptr);
  ln_kernel<<<8192, 256, 0, stream>>>(y, g2, be2, hb);
  gemm_bf16_kernel<<<g1536, 256, 0, stream>>>(hb, W1b, M, 1536, 384,
                                              MODE_RELU, b1, nullptr, nullptr,
                                              ab);
  gemm_bf16_kernel<<<g384, 256, 0, stream>>>(ab, W2b, M, 384, 1536, MODE_FRES,
                                             b2, x, (float*)d_out, nullptr);
}
